// IndependentRNNCritic_29437705846983
// MI455X (gfx1250) — matrix operand order repaired and checked by reading, not compiled, not run on hardware
//
#include <hip/hip_runtime.h>

// ---- problem constants --------------------------------------------------
#define BSZ 128
#define TT  128
#define AG  8
#define DD  128
#define HH  512
#define NR  1024            // BS*A rows per timestep
#define G3  1536            // 3*H
#define LPAD 16             // LDS row padding (bf16 elems), keeps 16B alignment

typedef __attribute__((ext_vector_type(8)))  float  v8f;
typedef __attribute__((ext_vector_type(16))) __bf16 v16bf;
typedef __attribute__((ext_vector_type(8)))  __bf16 v8bf;

__device__ __forceinline__ __bf16 f2bf(float f) { return (__bf16)f; }

// A-operand (16x32, MxK) from a row-major bf16 tile.
// ISA layout (16-bit A 16x32): lanes 0-15 rows M=0..15 hold K {0..7,16..23},
// lanes 16-31 hold K {8..15,24..31}; 2 elems per VGPR.
__device__ __forceinline__ v16bf ld_a(const __bf16* base, int stride, int lane, int k0) {
  const __bf16* r = base + (lane & 15) * stride + k0 + ((lane & 16) ? 8 : 0);
  v8bf lo = *(const v8bf*)(r);
  v8bf hi = *(const v8bf*)(r + 16);
  v16bf o;
#pragma unroll
  for (int i = 0; i < 8; ++i) { o[i] = lo[i]; o[i + 8] = hi[i]; }
  return o;
}

// B-operand (32x16, KxN) with B[k][n] = W[n0+n][k0+k], W row-major bf16:
// lane 0-15 -> K {0..7,16..23}, lane 16-31 -> K {8..15,24..31} (matches A).
__device__ __forceinline__ v16bf ld_b(const __bf16* wrow0, int stride, int lane, int k0) {
  const __bf16* r = wrow0 + (lane & 15) * stride + k0 + ((lane & 16) ? 8 : 0);
  v8bf lo = *(const v8bf*)(r);
  v8bf hi = *(const v8bf*)(r + 16);
  v16bf o;
#pragma unroll
  for (int i = 0; i < 8; ++i) { o[i] = lo[i]; o[i + 8] = hi[i]; }
  return o;
}

__device__ __forceinline__ v8f wmma_bf16(v16bf a, v16bf b, v8f c) {
  return __builtin_amdgcn_wmma_f32_16x16x32_bf16(false, a, false, b, (short)0, c, false, false);
}

__device__ __forceinline__ float sigm(float x) { return 1.f / (1.f + __expf(-x)); }
__device__ __forceinline__ float tanh_fast(float x) {
  float e = __expf(-2.f * x);
  return (1.f - e) / (1.f + e);
}

// ---- one-time prep: bf16 weight copies + zero h0 ------------------------
__global__ void rnncritic_prep(const float* __restrict__ W1,
                               const float* __restrict__ Wih,
                               const float* __restrict__ Whh,
                               __bf16* __restrict__ W1b,
                               __bf16* __restrict__ Wihb,
                               __bf16* __restrict__ Whhb,
                               float* __restrict__ h32,
                               __bf16* __restrict__ hbf) {
  size_t i0 = (size_t)blockIdx.x * blockDim.x + threadIdx.x;
  size_t gs = (size_t)gridDim.x * blockDim.x;
  for (size_t i = i0; i < (size_t)G3 * HH; i += gs) {
    Wihb[i] = f2bf(Wih[i]);
    Whhb[i] = f2bf(Whh[i]);
    if (i < (size_t)HH * DD)  W1b[i] = f2bf(W1[i]);
    if (i < (size_t)NR * HH) { h32[i] = 0.f; hbf[i] = f2bf(0.f); }
  }
}

// ---- fc1 + ReLU: x[t*1024+n][H] = relu(obs_row . W1^T + b1)  (bf16 out) --
__global__ __launch_bounds__(256) void rnncritic_fc1(const float* __restrict__ obs,
                                                     const __bf16* __restrict__ W1b,
                                                     const float* __restrict__ b1,
                                                     __bf16* __restrict__ xbf) {
  __shared__ __bf16 os[16][DD + LPAD];
  const int tid = threadIdx.x, wave = tid >> 5, lane = tid & 31;
  const int m0 = blockIdx.x * 16;       // global x-row base (t*1024 + n)
  {
    // stage+convert 16x128 fp32 obs tile -> bf16 LDS (one 8-elem chunk/thread)
    const int r = tid >> 4;
    const int k = (tid & 15) * 8;
    const int mg = m0 + r;
    const int t = mg >> 10, nl = mg & 1023, b = nl >> 3, a = nl & 7;
    const float* src = obs + (((size_t)(b * TT + t) * AG + a) * DD) + k;
    v8bf tmp;
#pragma unroll
    for (int i = 0; i < 8; ++i) tmp[i] = f2bf(src[i]);
    *(v8bf*)&os[r][k] = tmp;
  }
  __syncthreads();
  const int mbase = (lane & 16) ? 8 : 0;
#pragma unroll
  for (int j = 0; j < 4; ++j) {
    const int n0 = wave * 64 + j * 16;
    v8f acc = {0.f, 0.f, 0.f, 0.f, 0.f, 0.f, 0.f, 0.f};
#pragma unroll
    for (int k0 = 0; k0 < DD; k0 += 32)
      acc = wmma_bf16(ld_a(&os[0][0], DD + LPAD, lane, k0),
                      ld_b(W1b + (size_t)n0 * DD, DD, lane, k0), acc);
    const int ncol = n0 + (lane & 15);
    const float bias = b1[ncol];
#pragma unroll
    for (int e = 0; e < 8; ++e) {
      float v = acc[e] + bias;
      v = v > 0.f ? v : 0.f;                                   // ReLU
      xbf[(size_t)(m0 + mbase + e) * HH + ncol] = f2bf(v);
    }
  }
}

// ---- fused GRU step + value head (one launch per timestep) --------------
// WG = 16 rows x all 512 h-cols. 8 waves, each wave 4 N-tiles of 16.
__global__ __launch_bounds__(256) void rnncritic_step(const __bf16* __restrict__ xbf,
                                                      const __bf16* __restrict__ Wihb,
                                                      const __bf16* __restrict__ Whhb,
                                                      const float* __restrict__ b_ih,
                                                      const float* __restrict__ b_hh,
                                                      const float* __restrict__ hprev32,
                                                      const __bf16* __restrict__ hprevbf,
                                                      float* __restrict__ hnew32,
                                                      __bf16* __restrict__ hnewbf,
                                                      const float* __restrict__ Wv,
                                                      const float* __restrict__ bv,
                                                      float* __restrict__ out,
                                                      int t) {
  __shared__ __bf16 xs[16][HH + LPAD];
  __shared__ __bf16 hs[16][HH + LPAD];
  __shared__ float vtmp[256][8];
  const int tid = threadIdx.x, wave = tid >> 5, lane = tid & 31;
  const int m0 = blockIdx.x * 16;

  // stage x_t and h_prev (bf16, 16x512 each) into LDS
  const __bf16* xsrc = xbf + ((size_t)t * NR + m0) * HH;
  const __bf16* hsrc = hprevbf + (size_t)m0 * HH;
  for (int c = tid; c < 1024; c += 256) {           // 1024 chunks of 8 bf16
    const int r = c >> 6;
    const int k = (c & 63) * 8;
    *(v8bf*)&xs[r][k] = *(const v8bf*)(xsrc + (size_t)r * HH + k);
    *(v8bf*)&hs[r][k] = *(const v8bf*)(hsrc + (size_t)r * HH + k);
  }
  // prefetch next timestep's x rows into cache (global_prefetch_b8)
  if (t + 1 < TT) {
    const char* nx = (const char*)(xbf + ((size_t)(t + 1) * NR + m0) * HH);
    __builtin_prefetch(nx + tid * 64, 0, 1);
  }
  __syncthreads();

  const int mbase = (lane & 16) ? 8 : 0;
  float vpart[8] = {0.f, 0.f, 0.f, 0.f, 0.f, 0.f, 0.f, 0.f};

#pragma unroll 1
  for (int j = 0; j < 4; ++j) {
    const int n0 = wave * 64 + j * 16;
    v8f accr  = {0.f, 0.f, 0.f, 0.f, 0.f, 0.f, 0.f, 0.f};  // gi_r + gh_r
    v8f accz  = {0.f, 0.f, 0.f, 0.f, 0.f, 0.f, 0.f, 0.f};  // gi_z + gh_z
    v8f accni = {0.f, 0.f, 0.f, 0.f, 0.f, 0.f, 0.f, 0.f};  // gi_n
    v8f accnh = {0.f, 0.f, 0.f, 0.f, 0.f, 0.f, 0.f, 0.f};  // gh_n (gated by r)
#pragma unroll 4
    for (int k0 = 0; k0 < HH; k0 += 32) {
      v16bf ax = ld_a(&xs[0][0], HH + LPAD, lane, k0);
      v16bf ah = ld_a(&hs[0][0], HH + LPAD, lane, k0);
      accr  = wmma_bf16(ax, ld_b(Wihb + (size_t)(0 * HH + n0) * HH, HH, lane, k0), accr);
      accr  = wmma_bf16(ah, ld_b(Whhb + (size_t)(0 * HH + n0) * HH, HH, lane, k0), accr);
      accz  = wmma_bf16(ax, ld_b(Wihb + (size_t)(1 * HH + n0) * HH, HH, lane, k0), accz);
      accz  = wmma_bf16(ah, ld_b(Whhb + (size_t)(1 * HH + n0) * HH, HH, lane, k0), accz);
      accni = wmma_bf16(ax, ld_b(Wihb + (size_t)(2 * HH + n0) * HH, HH, lane, k0), accni);
      accnh = wmma_bf16(ah, ld_b(Whhb + (size_t)(2 * HH + n0) * HH, HH, lane, k0), accnh);
    }
    const int ncol = n0 + (lane & 15);
    const float br  = b_ih[ncol] + b_hh[ncol];
    const float bz  = b_ih[HH + ncol] + b_hh[HH + ncol];
    const float bin = b_ih[2 * HH + ncol];
    const float bhn = b_hh[2 * HH + ncol];
    const float wv  = Wv[ncol];
#pragma unroll
    for (int e = 0; e < 8; ++e) {
      const int mg = m0 + mbase + e;
      const size_t idx = (size_t)mg * HH + ncol;
      const float hp = hprev32[idx];
      const float r  = sigm(accr[e] + br);
      const float z  = sigm(accz[e] + bz);
      const float n  = tanh_fast(accni[e] + bin + r * (accnh[e] + bhn));
      const float hn = (1.f - z) * n + z * hp;
      hnew32[idx] = hn;
      hnewbf[idx] = f2bf(hn);
      vpart[e] += hn * wv;                          // value-head partial
    }
  }

  // deterministic value reduction: 256 lanes x 8 row-partials -> 16 rows
#pragma unroll
  for (int e = 0; e < 8; ++e) vtmp[tid][e] = vpart[e];
  __syncthreads();
  if (tid < 16) {
    const int eh = tid & 7;
    const int lbase = (tid >= 8) ? 16 : 0;
    float s = 0.f;
    for (int w = 0; w < 8; ++w)
#pragma unroll
      for (int l = 0; l < 16; ++l) s += vtmp[w * 32 + lbase + l][eh];
    const int nr = m0 + tid;
    const int b = nr >> 3, a = nr & 7;
    out[((size_t)b * TT + t) * AG + a] = s + bv[0];
  }
}

// ---- host launcher ------------------------------------------------------
extern "C" void kernel_launch(void* const* d_in, const int* in_sizes, int n_in,
                              void* d_out, int out_size, void* d_ws, size_t ws_size,
                              hipStream_t stream) {
  const float* obs  = (const float*)d_in[0];
  const float* W1   = (const float*)d_in[1];
  const float* b1   = (const float*)d_in[2];
  const float* W_ih = (const float*)d_in[3];
  const float* b_ih = (const float*)d_in[4];
  const float* W_hh = (const float*)d_in[5];
  const float* b_hh = (const float*)d_in[6];
  const float* Wv   = (const float*)d_in[7];
  const float* bv   = (const float*)d_in[8];
  float* out = (float*)d_out;

  char* ws = (char*)d_ws;
  size_t off = 0;
  auto carve = [&](size_t bytes) {
    char* p = ws + off;
    off += (bytes + 255) & ~(size_t)255;
    return p;
  };
  __bf16* xbf  = (__bf16*)carve((size_t)TT * NR * HH * 2);  // 128 MB activations
  __bf16* W1b  = (__bf16*)carve((size_t)HH * DD * 2);
  __bf16* Wihb = (__bf16*)carve((size_t)G3 * HH * 2);
  __bf16* Whhb = (__bf16*)carve((size_t)G3 * HH * 2);
  float*  hA32 = (float*)carve((size_t)NR * HH * 4);
  float*  hB32 = (float*)carve((size_t)NR * HH * 4);
  __bf16* hAbf = (__bf16*)carve((size_t)NR * HH * 2);
  __bf16* hBbf = (__bf16*)carve((size_t)NR * HH * 2);

  rnncritic_prep<<<512, 256, 0, stream>>>(W1, W_ih, W_hh, W1b, Wihb, Whhb, hA32, hAbf);
  rnncritic_fc1<<<(TT * NR) / 16, 256, 0, stream>>>(obs, W1b, b1, xbf);

  for (int t = 0; t < TT; ++t) {                     // ping-pong h buffers
    const float*  hp32 = (t & 1) ? hB32 : hA32;
    const __bf16* hpbf = (t & 1) ? hBbf : hAbf;
    float*  hn32 = (t & 1) ? hA32 : hB32;
    __bf16* hnbf = (t & 1) ? hAbf : hBbf;
    rnncritic_step<<<NR / 16, 256, 0, stream>>>(xbf, Wihb, Whhb, b_ih, b_hh,
                                                hp32, hpbf, hn32, hnbf,
                                                Wv, bv, out, t);
  }
}
